// GraphEncoder_31791347925669
// MI455X (gfx1250) — matrix operand order repaired and checked by reading, not compiled, not run on hardware
//
#include <hip/hip_runtime.h>
#include <math.h>

// ---- problem constants (match reference) ----
constexpr int NPG  = 512;          // nodes per graph
constexpr int DIM  = 128;          // feature dim
constexpr int NB   = 64;           // graphs (batch)
constexpr int NTOT = NB * NPG;     // 32768 nodes
constexpr int EPG  = 8192;         // edges per graph (E // B)
constexpr int ETOT = NTOT * 16;    // 524288 edges
constexpr int QF   = 32;           // feature chunk processed per pass
constexpr float EPSV = 1e-5f;

typedef __attribute__((ext_vector_type(16))) __bf16 v16bf;
typedef __attribute__((ext_vector_type(8)))  float  v8f;

union Frag16 { v16bf v; float4 f[2]; };

// =====================================================================
// init kernels
// =====================================================================
__global__ void k_init0(float* deg, float* stat) {
  int i = blockIdx.x * 256 + threadIdx.x;
  if (i < NTOT) deg[i] = 0.f;
  if (i < 512)  stat[i] = 0.f;       // sum[128] sq[128] mu[128] rsig[128]
}

__global__ void k_init1(const int* __restrict__ esrc, const int* __restrict__ edst,
                        float* deg, int* eloc) {
  int e = blockIdx.x * 256 + threadIdx.x;
  if (e >= ETOT) return;
  int g   = e / EPG;
  int dst = edst[e];
  atomicAdd(&deg[dst], 1.0f);
  int srcl = esrc[e] - g * NPG;      // local ids, each < 512
  int dstl = dst     - g * NPG;
  eloc[e] = srcl | (dstl << 16);
}

__global__ void k_init2(float* deg) { // deg -> dinv in place
  int i = blockIdx.x * 256 + threadIdx.x;
  if (i < NTOT) deg[i] = rsqrtf(1.0f + deg[i]);
}

// =====================================================================
// GCN conv layer: one block per graph, h resident in LDS.
//   u = dinv .* (bnrelu(h_prev) @ W)     (WMMA bf16 -> f32)
//   h_out[dst] = dinv[dst]*(sum_{e} u[src] + u[dst]) + b    (pre-BN)
// also accumulates global per-feature sum / sumsq for BN.
// =====================================================================
__global__ __launch_bounds__(256) void k_gcn(
    const float* __restrict__ x,       // layer 0 input [N,1]
    const float* __restrict__ h_prev,  // pre-BN features of prev layer
    const int*   __restrict__ eloc,    // packed local edges
    const float* __restrict__ dinv,
    const float* __restrict__ Wl,      // [128,128] (or W0 [1,128] for layer 0)
    const float* __restrict__ bl,      // [128]
    const float* __restrict__ gam,     // BN params of PREV layer (l>=1)
    const float* __restrict__ bet,
    const float* __restrict__ mu,      // BN stats of PREV layer
    const float* __restrict__ rsig,
    float* __restrict__ h_out,
    float* __restrict__ stat_sum,
    float* __restrict__ stat_sq,
    int layer)
{
  __shared__ __bf16 sh_h[NPG * DIM];     // 128 KB  bnrelu(h_prev) in bf16
  __shared__ __bf16 sh_wt[QF * DIM];     //   8 KB  W chunk, transposed [n][k]
  __shared__ float  sh_u[NPG * QF];      //  64 KB  dinv .* (h @ Wq)
  __shared__ float  sh_agg[NPG * QF];    //  64 KB  neighbor sums
  __shared__ float  sh_dinv[NPG];        //   2 KB
  __shared__ float  sh_x[NPG];           //   2 KB
  __shared__ int    sh_e[EPG];           //  32 KB  -> total 300 KB < 320 KB

  const int g    = blockIdx.x;
  const int tid  = threadIdx.x;
  const int lane = tid & 31;
  const int wv   = tid >> 5;

  for (int i = tid; i < NPG; i += 256) sh_dinv[i] = dinv[g * NPG + i];
  for (int i = tid; i < EPG; i += 256) sh_e[i]    = eloc[g * EPG + i];
  if (layer == 0) {
    for (int i = tid; i < NPG; i += 256) sh_x[i] = x[g * NPG + i];
  } else {
    for (int i = tid; i < NPG * DIM; i += 256) {
      int d = i & (DIM - 1);
      float v = h_prev[(size_t)g * NPG * DIM + i];
      v = gam[d] * (v - mu[d]) * rsig[d] + bet[d];
      v = v > 0.f ? v : 0.f;
      sh_h[i] = (__bf16)v;
    }
  }
  __syncthreads();

  for (int q = 0; q < 4; ++q) {
    // stage W chunk transposed; zero agg
    if (layer > 0) {
      for (int i = tid; i < QF * DIM; i += 256) {
        int n = i >> 7, k = i & (DIM - 1);
        sh_wt[n * DIM + k] = (__bf16)Wl[k * DIM + q * QF + n];
      }
    }
    for (int i = tid; i < NPG * QF; i += 256) sh_agg[i] = 0.f;
    __syncthreads();

    // ---- u = dinv .* (h @ Wq) ----
    if (layer == 0) {
      // D_IN == 1: hW[m][f] = x[m] * W0[f]
      for (int i = tid; i < NPG * QF; i += 256) {
        int n = i >> 5, f = i & (QF - 1);
        sh_u[i] = sh_dinv[n] * sh_x[n] * Wl[q * QF + f];
      }
    } else {
      const int grp = lane >> 4;   // half-wave group
      const int l15 = lane & 15;
      // 32 M-tiles x 2 N-tiles = 64 output tiles, 8 waves
      for (int t = wv; t < 64; t += 8) {
        int mt = t >> 1, nt = t & 1;
        v8f acc = {};
#pragma unroll
        for (int kb = 0; kb < 4; ++kb) {
          Frag16 fa, fb;
          // A 16x32 bf16: lane row = mt*16+l15; K = kb*32 + grp*8 + {0..7, 16..23}
          const float4* pa =
              (const float4*)&sh_h[(mt * 16 + l15) * DIM + kb * 32 + grp * 8];
          fa.f[0] = pa[0];
          fa.f[1] = pa[2];               // +16 elems = +32 bytes
          // B 32x16 bf16 from transposed W: lane col = nt*16+l15; K = kb*32 + grp*8 + {0..7, 16..23}
          const float4* pb =
              (const float4*)&sh_wt[(nt * 16 + l15) * DIM + kb * 32 + grp * 8];
          fb.f[0] = pb[0];
          fb.f[1] = pb[2];
          acc = __builtin_amdgcn_wmma_f32_16x16x32_bf16(
              false, fa.v, false, fb.v, (short)0, acc, false, false);
        }
#pragma unroll
        for (int r = 0; r < 8; ++r) {    // D layout: VGPR r -> M = r (+8 for hi half)
          int row = mt * 16 + grp * 8 + r;
          int col = nt * 16 + l15;
          sh_u[row * QF + col] = sh_dinv[row] * acc[r];
        }
      }
    }
    __syncthreads();

    // ---- edge aggregation: agg[dst] += u[src] ----
    {
      const int f  = tid & 31;   // 32 lanes = 32 consecutive feats -> bank-clean
      const int eo = tid >> 5;   // 8 edges in flight
      for (int eb = 0; eb < EPG; eb += 8) {
        int pe   = sh_e[eb + eo];
        int srcl = pe & 0xffff;
        int dstl = (pe >> 16) & 0xffff;
        atomicAdd(&sh_agg[dstl * QF + f], sh_u[srcl * QF + f]);
      }
    }
    __syncthreads();

    // ---- writeback (pre-BN) + BN statistics ----
    {
      const int f  = tid & 31;
      const int ng = tid >> 5;
      const float bb = bl[q * QF + f];
      float s = 0.f, sq = 0.f;
      for (int nb = 0; nb < NPG; nb += 8) {
        int n = nb + ng;
        float val = sh_dinv[n] * (sh_agg[n * QF + f] + sh_u[n * QF + f]) + bb;
        h_out[((size_t)g * NPG + n) * DIM + q * QF + f] = val;
        s += val; sq += val * val;
      }
      atomicAdd(&stat_sum[q * QF + f], s);
      atomicAdd(&stat_sq[q * QF + f], sq);
    }
    __syncthreads();
  }
}

// =====================================================================
// BN stats finalize (also the global sync point between layers)
// =====================================================================
__global__ void k_stats(float* stat) {
  int d = threadIdx.x;               // 128 threads
  float s = stat[d], q = stat[128 + d];
  float m = s * (1.0f / NTOT);
  float var = q * (1.0f / NTOT) - m * m;
  stat[256 + d] = m;
  stat[384 + d] = rsqrtf(var + EPSV);
  stat[d] = 0.f; stat[128 + d] = 0.f; // reset accumulators for next layer
}

// =====================================================================
// final BN+ReLU, attention pooling, dense output. One block per graph.
// =====================================================================
__global__ __launch_bounds__(256) void k_pool(
    const float* __restrict__ h_pre,
    const float* __restrict__ gam, const float* __restrict__ bet,
    const float* __restrict__ mu,  const float* __restrict__ rsig,
    const float* __restrict__ Watt,
    float* __restrict__ out)
{
  __shared__ float sh_h[NPG * DIM];   // 256 KB
  __shared__ float sh_red[256];
  __shared__ float sh_mean[DIM];
  __shared__ float sh_tg[DIM];
  __shared__ float sh_cf[NPG];
  __shared__ float sh_gf[DIM];

  const int g = blockIdx.x, tid = threadIdx.x;
  for (int i = tid; i < NPG * DIM; i += 256) {
    int n = i >> 7, d = i & 127;
    float v = h_pre[(size_t)g * NPG * DIM + i];
    v = gam[d] * (v - mu[d]) * rsig[d] + bet[d];
    v = v > 0.f ? v : 0.f;
    sh_h[i] = v;
    out[(size_t)g * NPG * 256 + n * 256 + d] = v;   // node_dense half
  }
  __syncthreads();
  // mean over nodes
  { int d = tid & 127, half = tid >> 7; float s = 0.f;
    for (int n = half * 256; n < half * 256 + 256; ++n) s += sh_h[n * DIM + d];
    sh_red[tid] = s; }
  __syncthreads();
  if (tid < DIM) sh_mean[tid] = (sh_red[tid] + sh_red[tid + 128]) * (1.0f / NPG);
  __syncthreads();
  // tg = tanh(mean @ W_att)
  if (tid < DIM) {
    float s = 0.f;
    for (int k = 0; k < DIM; ++k) s += sh_mean[k] * Watt[k * DIM + tid];
    sh_tg[tid] = tanhf(s);
  }
  __syncthreads();
  // coefs = sigmoid(10 * <h, tg>)
  for (int n = tid; n < NPG; n += 256) {
    float s = 0.f;
    for (int d = 0; d < DIM; ++d) s += sh_h[n * DIM + d] * sh_tg[d];
    sh_cf[n] = 1.0f / (1.0f + expf(-10.0f * s));
  }
  __syncthreads();
  // graph_feat = sum coefs[n] * h[n]
  { int d = tid & 127, half = tid >> 7; float s = 0.f;
    for (int n = half * 256; n < half * 256 + 256; ++n) s += sh_cf[n] * sh_h[n * DIM + d];
    sh_red[tid] = s; }
  __syncthreads();
  if (tid < DIM) sh_gf[tid] = sh_red[tid] + sh_red[tid + 128];
  __syncthreads();
  for (int i = tid; i < NPG * DIM; i += 256) {
    int n = i >> 7, d = i & 127;
    out[(size_t)g * NPG * 256 + n * 256 + 128 + d] = sh_gf[d];
  }
}

// =====================================================================
extern "C" void kernel_launch(void* const* d_in, const int* in_sizes, int n_in,
                              void* d_out, int out_size, void* d_ws, size_t ws_size,
                              hipStream_t stream) {
  const float* x    = (const float*)d_in[0];
  const int*   esrc = (const int*)d_in[1];
  const int*   edst = (const int*)d_in[2];
  /* d_in[3] = batch (implicit via layout, unused) */
  const float* W0   = (const float*)d_in[4];
  const float* b0   = (const float*)d_in[5];
  const float* Wr   = (const float*)d_in[6];
  const float* br   = (const float*)d_in[7];
  const float* gamma= (const float*)d_in[8];
  const float* beta = (const float*)d_in[9];
  const float* Watt = (const float*)d_in[10];
  float* out = (float*)d_out;

  char* ws = (char*)d_ws;
  float* dinv = (float*)ws;  ws += (size_t)NTOT * sizeof(float);
  int*   eloc = (int*)ws;    ws += (size_t)ETOT * sizeof(int);
  float* stat = (float*)ws;  ws += 512 * sizeof(float);
  float* hA   = (float*)ws;  ws += (size_t)NTOT * DIM * sizeof(float);
  float* hB   = (float*)ws;

  k_init0<<<(NTOT + 255) / 256, 256, 0, stream>>>(dinv, stat);
  k_init1<<<(ETOT + 255) / 256, 256, 0, stream>>>(esrc, edst, dinv, eloc);
  k_init2<<<(NTOT + 255) / 256, 256, 0, stream>>>(dinv);

  for (int l = 0; l < 10; ++l) {
    const float* Wl  = (l == 0) ? W0 : Wr + (size_t)(l - 1) * DIM * DIM;
    const float* bl  = (l == 0) ? b0 : br + (size_t)(l - 1) * DIM;
    const float* gam = gamma + (size_t)(l ? l - 1 : 0) * DIM;  // prev-layer BN
    const float* bet = beta  + (size_t)(l ? l - 1 : 0) * DIM;
    float*       ho  = (l & 1) ? hB : hA;
    const float* hi  = (l & 1) ? hA : hB;
    k_gcn<<<NB, 256, 0, stream>>>(x, hi, eloc, dinv, Wl, bl, gam, bet,
                                  stat + 256, stat + 384, ho, stat, stat + 128, l);
    k_stats<<<1, 128, 0, stream>>>(stat);  // global sync + BN finalize
  }
  k_pool<<<NB, 256, 0, stream>>>(hB, gamma + 9 * DIM, beta + 9 * DIM,
                                 stat + 256, stat + 384, Watt, out);
}
